// TransformerLayerRef_39067022524543
// MI455X (gfx1250) — matrix operand order repaired and checked by reading, not compiled, not run on hardware
//
#include <hip/hip_runtime.h>
#include <cstdint>
#include <cstddef>

typedef unsigned short u16;
typedef __attribute__((ext_vector_type(16))) __bf16 bf16x16;
typedef __attribute__((ext_vector_type(8)))  float  f32x8;
typedef __attribute__((ext_vector_type(4)))  unsigned int u32x4;
typedef __attribute__((ext_vector_type(8)))  int i32x8;
typedef __attribute__((ext_vector_type(4)))  int i32x4;

#define TT  2048
#define DD  2048
#define NH  32
#define NKV 8
#define DHD 64
#define II  5632

union FragBF {
  bf16x16 v;
  uint4   u[2];
};

__device__ __forceinline__ u16 f2bf(float f) {
  unsigned u = __float_as_uint(f);
  unsigned r = (u + 0x7FFFu + ((u >> 16) & 1u)) >> 16;   // round-to-nearest-even
  return (u16)r;
}
__device__ __forceinline__ float bf2f(u16 h) {
  return __uint_as_float(((unsigned)h) << 16);
}
__device__ __forceinline__ f32x8 zero8() {
  f32x8 z;
#pragma unroll
  for (int i = 0; i < 8; ++i) z[i] = 0.f;
  return z;
}
__device__ __forceinline__ f32x8 wmma_bf16(FragBF a, FragBF b, f32x8 c) {
  return __builtin_amdgcn_wmma_f32_16x16x32_bf16(false, a.v, false, b.v,
                                                 (short)0, c, false, false);
}

// ---------------------------------------------------------------------------
// Tensor Data Mover: async 2-D tile load global -> LDS (D# per ISA ch.8).
//   flags word = group1[31:0]: data_size=2B (bit16), optional LDS padding.
//   2-D tile: tile_dim0 elements along dim0 (contiguous), tile_dim1 rows,
//   row stride = stride0 elements. Groups 2/3 zero (tile_dim3/4 unused).
// ---------------------------------------------------------------------------
#define TDM_FLAGS_PLAIN      0x00010000u                         // data_size=2B
#define TDM_FLAGS_PAD16_4    (0x00010000u | (1u << 20) |         /* pad_enable */ \
                              (3u << 22) |                       /* every 16 DW */ \
                              (3u << 25))                        /* pad 4 DW    */

__device__ __forceinline__ void tdm_load_2d(unsigned ldsOff, const void* gptr,
                                            unsigned td0, unsigned td1,
                                            unsigned long long stride0,
                                            unsigned tile0, unsigned tile1,
                                            unsigned flags) {
  unsigned long long ga = (unsigned long long)(uintptr_t)gptr;
  u32x4 g0;
  g0[0] = 1u;                                            // count=1, user desc
  g0[1] = ldsOff;                                        // lds_addr (bytes)
  g0[2] = (unsigned)ga;                                  // global_addr[31:0]
  g0[3] = (unsigned)((ga >> 32) & 0x01FFFFFFull) | (2u << 30);  // [56:32]|type=2
  i32x8 g1;
  g1[0] = (int)flags;                                    // mask=0|data_size|pad
  g1[1] = (int)((td0 & 0xFFFFu) << 16);                  // tensor_dim0[15:0]
  g1[2] = (int)(((td0 >> 16) & 0xFFFFu) | ((td1 & 0xFFFFu) << 16));
  g1[3] = (int)(((td1 >> 16) & 0xFFFFu) | (tile0 << 16));// tile_dim0
  g1[4] = (int)(tile1 & 0xFFFFu);                        // tile_dim1, tile_dim2=0
  g1[5] = (int)(unsigned)(stride0 & 0xFFFFFFFFull);      // dim0_stride[31:0]
  g1[6] = (int)(unsigned)((stride0 >> 32) & 0xFFFFull);  // dim0_stride[47:32]
  g1[7] = 0;                                             // dim1_stride unused
  i32x4 gz;
  gz[0] = gz[1] = gz[2] = gz[3] = 0;
#if defined(__clang_major__) && (__clang_major__ >= 23)
  i32x8 gz8;
#pragma unroll
  for (int i = 0; i < 8; ++i) gz8[i] = 0;
  __builtin_amdgcn_tensor_load_to_lds(g0, g1, gz, gz, gz8, 0);
#else
  __builtin_amdgcn_tensor_load_to_lds(g0, g1, gz, gz, 0);
#endif
}

// ---------------------------------------------------------------------------
// RMSNorm: fp32 row -> bf16 row (x * rsqrt(mean(x^2)+eps) * w)
// ---------------------------------------------------------------------------
__global__ __launch_bounds__(256)
void k_rmsnorm(const float* __restrict__ X, const float* __restrict__ W,
               u16* __restrict__ O) {
  const int row = blockIdx.x;
  const int tid = threadIdx.x;
  const float4* xr = (const float4*)(X + (size_t)row * DD);
  float4 a = xr[tid];
  float4 b = xr[tid + 256];
  float ss = a.x*a.x + a.y*a.y + a.z*a.z + a.w*a.w
           + b.x*b.x + b.y*b.y + b.z*b.z + b.w*b.w;
#pragma unroll
  for (int m = 1; m <= 16; m <<= 1) ss += __shfl_xor(ss, m, 32);
  __shared__ float red[8];
  __shared__ float s_scale;
  if ((tid & 31) == 0) red[tid >> 5] = ss;
  __syncthreads();
  if (tid == 0) {
    float t = 0.f;
#pragma unroll
    for (int i = 0; i < 8; ++i) t += red[i];
    s_scale = rsqrtf(t / (float)DD + 1e-5f);
  }
  __syncthreads();
  const float sc = s_scale;
  const float4* wr = (const float4*)W;
  float4 wa = wr[tid], wb = wr[tid + 256];
  uint2 o0, o1;
  o0.x = (unsigned)f2bf(a.x*sc*wa.x) | ((unsigned)f2bf(a.y*sc*wa.y) << 16);
  o0.y = (unsigned)f2bf(a.z*sc*wa.z) | ((unsigned)f2bf(a.w*sc*wa.w) << 16);
  o1.x = (unsigned)f2bf(b.x*sc*wb.x) | ((unsigned)f2bf(b.y*sc*wb.y) << 16);
  o1.y = (unsigned)f2bf(b.z*sc*wb.z) | ((unsigned)f2bf(b.w*sc*wb.w) << 16);
  *(uint2*)(O + (size_t)row * DD + tid * 4)        = o0;
  *(uint2*)(O + (size_t)row * DD + (tid + 256) * 4) = o1;
}

// ---------------------------------------------------------------------------
// GEMM: C[M,N] = A(bf16, row-major, lda) x W(fp32, row-major KxN, converted
// to bf16 on the fly). 128x128 block tile, K-step 32, 8 waves (4x2).
// A tile (128x32 bf16, 80B padded rows) is DMA'd by the TDM (wave 0) while
// all 256 threads stream + convert the fp32 B tile -> overlap of the two
// staging streams; wave 0 drains TENSORcnt before the publishing barrier.
// ---------------------------------------------------------------------------
template <bool ADD_RES, bool OUT_BF16>
__global__ __launch_bounds__(256)
void k_gemm(const u16* __restrict__ A, int lda,
            const float* __restrict__ W, int ldw,
            const float* __restrict__ Res,
            void* __restrict__ Cout, int N, int K, float oscale) {
  __shared__ u16 As[128 * 40];   // [row][k], row stride 40 (80B = 64B + 16B pad)
  __shared__ u16 Bs[128 * 40];   // [n][k]   (K-contiguous per output column)

  const int tid  = threadIdx.x;
  const int lane = tid & 31;
  const int wave = tid >> 5;
  const int wm   = wave >> 1;              // 0..3 : wave M tile
  const int wn   = wave & 1;               // 0..1 : wave N tile
  const int row0 = blockIdx.y * 128;
  const int n0   = blockIdx.x * 128;
  const unsigned asOff = (unsigned)(uintptr_t)(void*)As;

  f32x8 acc[2][4];
#pragma unroll
  for (int mi = 0; mi < 2; ++mi)
#pragma unroll
    for (int ni = 0; ni < 4; ++ni) acc[mi][ni] = zero8();

  const int bk  = tid >> 3;               // B-stage k row 0..31
  const int bng = tid & 7;                // B-stage n group (16 cols each)

  for (int k0 = 0; k0 < K; k0 += 32) {
    // A tile via Tensor Data Mover (issued once, by wave 0)
    if (wave == 0) {
      tdm_load_2d(asOff, A + (size_t)row0 * lda + k0,
                  (unsigned)lda, 1u << 20, (unsigned long long)lda,
                  32u, 128u, TDM_FLAGS_PAD16_4);
    }
    // B tile (32x128 fp32 -> bf16, transposed to [n][k]) by all threads
    {
      const float4* wv4 = (const float4*)(W + (size_t)(k0 + bk) * ldw + n0 + bng * 16);
#pragma unroll
      for (int q = 0; q < 4; ++q) {
        float4 f = wv4[q];
        int nb = bng * 16 + q * 4;
        Bs[(nb + 0) * 40 + bk] = f2bf(f.x);
        Bs[(nb + 1) * 40 + bk] = f2bf(f.y);
        Bs[(nb + 2) * 40 + bk] = f2bf(f.z);
        Bs[(nb + 3) * 40 + bk] = f2bf(f.w);
      }
    }
    if (wave == 0) __builtin_amdgcn_s_wait_tensorcnt((short)0);
    __syncthreads();

    FragBF af[2], bfr[4];
    const int ao = (lane < 16) ? 0 : 16;   // A: K0-7/K16-23 vs K8-15/K24-31
#pragma unroll
    for (int mi = 0; mi < 2; ++mi) {
      const char* base = (const char*)As + (wm * 32 + mi * 16 + (lane & 15)) * 80;
      af[mi].u[0] = *(const uint4*)(base + ao);
      af[mi].u[1] = *(const uint4*)(base + ao + 32);
    }
    const int bo = (lane < 16) ? 0 : 16;   // B: K0-7/K16-23 vs K8-15/K24-31
#pragma unroll
    for (int ni = 0; ni < 4; ++ni) {
      const char* base = (const char*)Bs + (wn * 64 + ni * 16 + (lane & 15)) * 80;
      bfr[ni].u[0] = *(const uint4*)(base + bo);
      bfr[ni].u[1] = *(const uint4*)(base + bo + 32);
    }
#pragma unroll
    for (int mi = 0; mi < 2; ++mi)
#pragma unroll
      for (int ni = 0; ni < 4; ++ni)
        acc[mi][ni] = wmma_bf16(af[mi], bfr[ni], acc[mi][ni]);
    __syncthreads();
  }

  // epilogue: C layout element (M = r + 8*(lane>=16), N = lane&15)
  const int rhalf = (lane < 16) ? 0 : 8;
  const int cl    = lane & 15;
#pragma unroll
  for (int mi = 0; mi < 2; ++mi)
#pragma unroll
    for (int ni = 0; ni < 4; ++ni)
#pragma unroll
      for (int rr = 0; rr < 8; ++rr) {
        int row = row0 + wm * 32 + mi * 16 + rr + rhalf;
        int col = n0 + wn * 64 + ni * 16 + cl;
        size_t idx = (size_t)row * N + col;
        float v = acc[mi][ni][rr] * oscale;
        if (ADD_RES) v += Res[idx];
        if (OUT_BF16) ((u16*)Cout)[idx] = f2bf(v);
        else          ((float*)Cout)[idx] = v;
      }
}

// ---------------------------------------------------------------------------
// Flash attention, causal, GQA (4 Q heads per KV head).
// Block = 1 head x 128 Q rows; 8 waves, each owns a 16-row strip.
// K chunk (32x64 bf16) is DMA'd by the TDM; V chunk is transposed manually
// (TDM is a raw mover). Scale 1/sqrt(DH) pre-folded into Q.
// ---------------------------------------------------------------------------
__global__ __launch_bounds__(256)
void k_attn(const u16* __restrict__ Q, const u16* __restrict__ Kb,
            const u16* __restrict__ Vb, u16* __restrict__ O) {
  __shared__ u16 Ks[32 * 64];        // [s][dh]   row-major (K^T cols contiguous)
  __shared__ u16 Vt[64 * 32];        // [dh][s]   (V cols contiguous for PV B-frag)
  __shared__ u16 Ps[8 * 16 * 32];    // per-wave P scratch (16x32 bf16)

  const int tid  = threadIdx.x;
  const int lane = tid & 31;
  const int wave = tid >> 5;
  const int q0   = blockIdx.x * 128;
  const int head = blockIdx.y;
  const int kvh  = head >> 2;
  const int rowBase = q0 + wave * 16;
  const int myrow   = rowBase + (lane & 15);
  const unsigned ksOff = (unsigned)(uintptr_t)(void*)Ks;

  // Q fragments (16x64 -> two 16x32 A-frags), loaded straight from global
  FragBF qf0, qf1;
  {
    const char* qb = (const char*)(Q + (size_t)myrow * (NH * DHD) + head * DHD);
    const int qo = (lane < 16) ? 0 : 16;
    qf0.u[0] = *(const uint4*)(qb + qo);
    qf0.u[1] = *(const uint4*)(qb + 32 + qo);
    qf1.u[0] = *(const uint4*)(qb + 64 + qo);
    qf1.u[1] = *(const uint4*)(qb + 96 + qo);
  }

  f32x8 oacc[4];
#pragma unroll
  for (int ni = 0; ni < 4; ++ni) oacc[ni] = zero8();
  float mrow[8], lrow[8];
#pragma unroll
  for (int i = 0; i < 8; ++i) { mrow[i] = -1e30f; lrow[i] = 0.f; }

  const int numChunks = q0 / 32 + 4;         // cover rows q0 .. q0+127 causally
  const int vt_s = tid & 31, vt_g = tid >> 5;

  for (int c = 0; c < numChunks; ++c) {
    const int s0 = c * 32;
    // stage K chunk via TDM (32 rows x 64 dh, row stride 512 elements)
    if (wave == 0) {
      tdm_load_2d(ksOff, Kb + (size_t)s0 * (NKV * DHD) + kvh * DHD,
                  (unsigned)(NKV * DHD), 1u << 20,
                  (unsigned long long)(NKV * DHD),
                  64u, 32u, TDM_FLAGS_PLAIN);
    }
    // stage V chunk transposed -> Vt[dh][s] (all threads)
    {
      const u16* vsrc = Vb + (size_t)(s0 + vt_s) * (NKV * DHD) + kvh * DHD + vt_g * 8;
#pragma unroll
      for (int i = 0; i < 8; ++i) Vt[(vt_g * 8 + i) * 32 + vt_s] = vsrc[i];
    }
    if (wave == 0) __builtin_amdgcn_s_wait_tensorcnt((short)0);
    __syncthreads();

    if (s0 <= rowBase + 15) {                // wave-uniform: EXEC stays all-1s
      // S = Q @ K^T : two 16x16 tiles (cols s0..15, s16..31)
      f32x8 sacc[2];
      sacc[0] = zero8(); sacc[1] = zero8();
      const int bo = (lane < 16) ? 0 : 16;
#pragma unroll
      for (int j = 0; j < 2; ++j) {
        const char* kbase = (const char*)Ks + (j * 16 + (lane & 15)) * 128;
        FragBF b0, b1;
        b0.u[0] = *(const uint4*)(kbase + bo);
        b0.u[1] = *(const uint4*)(kbase + bo + 32);
        b1.u[0] = *(const uint4*)(kbase + 64 + bo);
        b1.u[1] = *(const uint4*)(kbase + 64 + bo + 32);
        sacc[j] = wmma_bf16(qf0, b0, sacc[j]);
        sacc[j] = wmma_bf16(qf1, b1, sacc[j]);
      }

      // online softmax over the 32 new columns
      const int rhalf = (lane < 16) ? 0 : 8;
      const int cl    = lane & 15;
      u16* pp = Ps + wave * 512;
#pragma unroll
      for (int rr = 0; rr < 8; ++rr) {
        const int rowg = rowBase + rr + rhalf;
        float sv0 = sacc[0][rr];
        float sv1 = sacc[1][rr];
        if (s0 + cl > rowg)       sv0 = -1e30f;     // causal mask
        if (s0 + 16 + cl > rowg)  sv1 = -1e30f;
        float mx = fmaxf(sv0, sv1);
#pragma unroll
        for (int m = 1; m <= 8; m <<= 1) mx = fmaxf(mx, __shfl_xor(mx, m, 32));
        const float mnew  = fmaxf(mrow[rr], mx);
        const float alpha = __expf(mrow[rr] - mnew);
        const float p0 = (sv0 > -1e29f) ? __expf(sv0 - mnew) : 0.f;
        const float p1 = (sv1 > -1e29f) ? __expf(sv1 - mnew) : 0.f;
        float ls = p0 + p1;
#pragma unroll
        for (int m = 1; m <= 8; m <<= 1) ls += __shfl_xor(ls, m, 32);
        lrow[rr] = lrow[rr] * alpha + ls;
        mrow[rr] = mnew;
#pragma unroll
        for (int ni = 0; ni < 4; ++ni) oacc[ni][rr] *= alpha;
        const int prow = rr + rhalf;
        pp[prow * 32 + cl]      = f2bf(p0);
        pp[prow * 32 + 16 + cl] = f2bf(p1);
      }
      asm volatile("s_wait_dscnt 0x0" ::: "memory");

      // reload P as an A-fragment (16x32)
      FragBF pf;
      {
        const char* pb = (const char*)pp + (lane & 15) * 64;
        const int po = (lane < 16) ? 0 : 16;
        pf.u[0] = *(const uint4*)(pb + po);
        pf.u[1] = *(const uint4*)(pb + po + 32);
      }
      // PV: four 32x16 B-frags from Vt
      const int vo = (lane < 16) ? 0 : 16;
#pragma unroll
      for (int ni = 0; ni < 4; ++ni) {
        const char* vbase = (const char*)Vt + (ni * 16 + (lane & 15)) * 64;
        FragBF vf;
        vf.u[0] = *(const uint4*)(vbase + vo);
        vf.u[1] = *(const uint4*)(vbase + vo + 32);
        oacc[ni] = wmma_bf16(pf, vf, oacc[ni]);
      }
    }
    __syncthreads();
  }

  // normalize and store bf16 output
  const int rhalf = (lane < 16) ? 0 : 8;
  const int cl    = lane & 15;
#pragma unroll
  for (int ni = 0; ni < 4; ++ni)
#pragma unroll
    for (int rr = 0; rr < 8; ++rr) {
      const int rowg = rowBase + rr + rhalf;
      const float v = oacc[ni][rr] / lrow[rr];
      O[(size_t)rowg * (NH * DHD) + head * DHD + ni * 16 + cl] = f2bf(v);
    }
}

// ---------------------------------------------------------------------------
// SwiGLU: fc1 is [T, 2I] bf16; write silu(gate)*val into first I columns.
// ---------------------------------------------------------------------------
__global__ __launch_bounds__(256)
void k_swiglu(u16* __restrict__ FC) {
  const size_t idx = (size_t)blockIdx.x * 256 + threadIdx.x;
  const int t = (int)(idx / II);
  const int i = (int)(idx % II);
  const size_t base = (size_t)t * (2 * II);
  const float g = bf2f(FC[base + i]);
  const float v = bf2f(FC[base + II + i]);
  const float s = g / (1.f + __expf(-g)) * v;
  FC[base + i] = f2bf(s);
}

// ---------------------------------------------------------------------------
extern "C" void kernel_launch(void* const* d_in, const int* in_sizes, int n_in,
                              void* d_out, int out_size, void* d_ws, size_t ws_size,
                              hipStream_t stream) {
  (void)in_sizes; (void)n_in; (void)out_size; (void)ws_size;
  const float* x   = (const float*)d_in[0];
  const float* ln1 = (const float*)d_in[1];
  const float* ln2 = (const float*)d_in[2];
  const float* wq  = (const float*)d_in[3];
  const float* wk  = (const float*)d_in[4];
  const float* wv  = (const float*)d_in[5];
  const float* wo  = (const float*)d_in[6];
  const float* w1  = (const float*)d_in[7];
  const float* w2  = (const float*)d_in[8];
  float* out = (float*)d_out;

  char* ws = (char*)d_ws;
  size_t off = 0;
  auto alloc = [&](size_t bytes) {
    char* p = ws + off;
    off += (bytes + 255) & ~(size_t)255;
    return p;
  };
  u16*   h1   = (u16*)  alloc((size_t)TT * DD * 2);          // rmsnorm1 (bf16)
  u16*   qb   = (u16*)  alloc((size_t)TT * NH  * DHD * 2);   // Q (scaled, bf16)
  u16*   kb   = (u16*)  alloc((size_t)TT * NKV * DHD * 2);   // K (bf16)
  u16*   vb   = (u16*)  alloc((size_t)TT * NKV * DHD * 2);   // V (bf16)
  u16*   ab   = (u16*)  alloc((size_t)TT * NH  * DHD * 2);   // attn out (bf16)
  float* xmid = (float*)alloc((size_t)TT * DD * 4);          // x + attn@wo (fp32)
  u16*   h2   = (u16*)  alloc((size_t)TT * DD * 2);          // rmsnorm2 (bf16)
  u16*   fc1  = (u16*)  alloc((size_t)TT * 2 * II * 2);      // fc1 / swiglu (bf16)

  k_rmsnorm<<<TT, 256, 0, stream>>>(x, ln1, h1);

  k_gemm<false, true><<<dim3((NH * DHD) / 128, TT / 128), 256, 0, stream>>>(
      h1, DD, wq, NH * DHD, nullptr, qb, NH * DHD, DD, 0.125f);  // 1/sqrt(64)
  k_gemm<false, true><<<dim3((NKV * DHD) / 128, TT / 128), 256, 0, stream>>>(
      h1, DD, wk, NKV * DHD, nullptr, kb, NKV * DHD, DD, 1.f);
  k_gemm<false, true><<<dim3((NKV * DHD) / 128, TT / 128), 256, 0, stream>>>(
      h1, DD, wv, NKV * DHD, nullptr, vb, NKV * DHD, DD, 1.f);

  k_attn<<<dim3(TT / 128, NH), 256, 0, stream>>>(qb, kb, vb, ab);

  k_gemm<true, false><<<dim3(DD / 128, TT / 128), 256, 0, stream>>>(
      ab, NH * DHD, wo, DD, x, xmid, DD, NH * DHD, 1.f);

  k_rmsnorm<<<TT, 256, 0, stream>>>(xmid, ln2, h2);

  k_gemm<false, true><<<dim3((2 * II) / 128, TT / 128), 256, 0, stream>>>(
      h2, DD, w1, 2 * II, nullptr, fc1, 2 * II, DD, 1.f);

  k_swiglu<<<(unsigned)(((size_t)TT * II) / 256), 256, 0, stream>>>(fc1);

  k_gemm<true, false><<<dim3(DD / 128, TT / 128), 256, 0, stream>>>(
      fc1, 2 * II, w2, DD, xmid, out, DD, II, 1.f);
}
